// TernaryLinear_61400852463755
// MI455X (gfx1250) — 8-bit matrix operand order repaired and checked by reading, compiled, not run on hardware
//
#include <hip/hip_runtime.h>

typedef int   v16i __attribute__((ext_vector_type(16)));
typedef float v8f  __attribute__((ext_vector_type(8)));
typedef int   v4i_vs __attribute__((vector_size(16)));

#define T_DIM 16384
#define DIN   2048
#define DOUT  2048

#define BM 128
#define BN 128
#define BK 128
#define LDSK 144
#define TILE_BYTES (BM * LDSK)

struct __align__(16) V16 { unsigned long long a, b; };

union FragU {
    v16i v;
    unsigned long long u64[8];
};

#ifndef __has_builtin
#define __has_builtin(x) 0
#endif

#if __has_builtin(__builtin_amdgcn_global_load_async_to_lds_b128)
#define HAVE_ASYNC_LDS 1
#else
#define HAVE_ASYNC_LDS 0
#endif

typedef __attribute__((address_space(1))) v4i_vs gmem_v4i;
typedef __attribute__((address_space(3))) v4i_vs lds_v4i;

__device__ __forceinline__ void copy16_to_lds(const signed char* g, signed char* l)
{
#if HAVE_ASYNC_LDS
    __builtin_amdgcn_global_load_async_to_lds_b128(
        (gmem_v4i*)(unsigned long long)g,
        (lds_v4i*)(unsigned int)(unsigned long long)l,
         0,  0);
#else
    *reinterpret_cast<V16*>(l) = *reinterpret_cast<const V16*>(g);
#endif
}

#if HAVE_ASYNC_LDS
# if __has_builtin(__builtin_amdgcn_s_wait_asynccnt)
#  define WAIT_ASYNC(n) __builtin_amdgcn_s_wait_asynccnt(n)
# else
#  define WAIT_ASYNC(n) asm volatile("s_wait_asynccnt %0" :: "i"(n) : "memory")
# endif
#else
# define WAIT_ASYNC(n) ((void)0)
#endif

__device__ __forceinline__ unsigned fp8_ternary(float x)
{
    const float r = fmaxf(-1.0f, fminf(1.0f, rintf(x)));
    return (r == 0.0f) ? 0x00u : ((r > 0.0f) ? 0x38u : 0xB8u);
}

__global__ __launch_bounds__(256) void quantize_fp8_kernel(
    const float* __restrict__ x, unsigned char* __restrict__ q, int n4)
{
    int i = blockIdx.x * blockDim.x + threadIdx.x;
    if (i >= n4) return;
    const float4 f = reinterpret_cast<const float4*>(x)[i];
    unsigned packed = fp8_ternary(f.x)
                    | (fp8_ternary(f.y) << 8)
                    | (fp8_ternary(f.z) << 16)
                    | (fp8_ternary(f.w) << 24);
    reinterpret_cast<unsigned*>(q)[i] = packed;
}

__global__ __launch_bounds__(256) void ternary_wmma_fp8_kernel(
    const signed char* __restrict__ Xq,
    const signed char* __restrict__ Wq,
    const float* __restrict__ bias,
    float* __restrict__ out)
{
    __shared__ __align__(16) signed char As[2][TILE_BYTES];
    __shared__ __align__(16) signed char Bs[2][TILE_BYTES];

    const int tid    = threadIdx.x;
    const int lane   = tid & 31;
    const int wave   = tid >> 5;
    const int wave_m = wave & 3;
    const int wave_n = wave >> 2;

    const int block_m = blockIdx.y * BM;
    const int block_n = blockIdx.x * BN;

    const int lh  = lane >> 4;
    const int l15 = lane & 15;

    const int st_row = tid >> 3;
    const int st_col = (tid & 7) * 16;

    auto stage = [&](int kt, int buf) {
        const int kbase = kt * BK;
        #pragma unroll
        for (int p = 0; p < 4; ++p) {
            const int row = st_row + p * 32;
            copy16_to_lds(Xq + (size_t)(block_m + row) * DIN + kbase + st_col,
                          &As[buf][row * LDSK + st_col]);
            copy16_to_lds(Wq + (size_t)(block_n + row) * DIN + kbase + st_col,
                          &Bs[buf][row * LDSK + st_col]);
        }
    };

    v8f acc[2][4] = {};

    const int kTiles = DIN / BK;
    stage(0, 0);

    for (int kt = 0; kt < kTiles; ++kt) {
        const int cur = kt & 1;
        if (kt + 1 < kTiles) {
            stage(kt + 1, cur ^ 1);
            WAIT_ASYNC(8);
        } else {
            WAIT_ASYNC(0);
        }
        __syncthreads();

        FragU afrag[2];
        #pragma unroll
        for (int mt = 0; mt < 2; ++mt) {
            const signed char* arow =
                &As[cur][(wave_m * 32 + mt * 16 + l15) * LDSK];
            #pragma unroll
            for (int p = 0; p < 4; ++p) {
                afrag[mt].u64[p]     = *reinterpret_cast<const unsigned long long*>(
                    arow + (p >> 1) * 32 + lh * 16 + (p & 1) * 8);
                afrag[mt].u64[4 + p] = *reinterpret_cast<const unsigned long long*>(
                    arow + 64 + (p >> 1) * 32 + lh * 16 + (p & 1) * 8);
            }
        }
        FragU bfrag[4];
        #pragma unroll
        for (int nt = 0; nt < 4; ++nt) {
            const signed char* brow =
                &Bs[cur][(wave_n * 64 + nt * 16 + l15) * LDSK];
            #pragma unroll
            for (int g = 0; g < 4; ++g) {
                bfrag[nt].u64[2 * g]     = *reinterpret_cast<const unsigned long long*>(
                    brow + g * 32 + lh * 16);
                bfrag[nt].u64[2 * g + 1] = *reinterpret_cast<const unsigned long long*>(
                    brow + g * 32 + lh * 16 + 8);
            }
        }

        #pragma unroll
        for (int mt = 0; mt < 2; ++mt)
            #pragma unroll
            for (int nt = 0; nt < 4; ++nt)
                acc[mt][nt] = __builtin_amdgcn_wmma_f32_16x16x128_fp8_fp8(
                    afrag[mt].v, bfrag[nt].v, (short)0, acc[mt][nt],
                     false,  false);

        __syncthreads();
    }

    #pragma unroll
    for (int mt = 0; mt < 2; ++mt) {
        #pragma unroll
        for (int nt = 0; nt < 4; ++nt) {
            const int col = block_n + wave_n * 64 + nt * 16 + l15;
            const float b = bias[col];
            #pragma unroll
            for (int v = 0; v < 8; ++v) {
                const int row = block_m + wave_m * 32 + mt * 16 + v + 8 * lh;
                out[(size_t)row * DOUT + col] = acc[mt][nt][v] + b;
            }
        }
    }
}

extern "C" void kernel_launch(void* const* d_in, const int* in_sizes, int n_in,
                              void* d_out, int out_size, void* d_ws, size_t ws_size,
                              hipStream_t stream)
{
    const float* x    = (const float*)d_in[0];
    const float* w    = (const float*)d_in[1];
    const float* bias = (const float*)d_in[2];
    float* out        = (float*)d_out;

    unsigned char* xq = (unsigned char*)d_ws;
    unsigned char* wq = xq + (size_t)T_DIM * DIN;

    {
        const int n4 = T_DIM * DIN / 4;
        quantize_fp8_kernel<<<n4 / 256, 256, 0, stream>>>(x, xq, n4);
    }
    {
        const int n4 = DOUT * DIN / 4;
        quantize_fp8_kernel<<<n4 / 256, 256, 0, stream>>>(w, wq, n4);
    }

    dim3 grid(DOUT / BN, T_DIM / BM);
    ternary_wmma_fp8_kernel<<<grid, 256, 0, stream>>>(
        (const signed char*)xq, (const signed char*)wq, bias, out);
}
